// FP8Linear_13116830122041
// MI455X (gfx1250) — 8-bit matrix operand order repaired and checked by reading, compiled, not run on hardware
//
#include <hip/hip_runtime.h>
#include <stdint.h>

typedef __attribute__((ext_vector_type(16))) int          v16i;
typedef __attribute__((ext_vector_type(8)))  float        v8f;
typedef __attribute__((ext_vector_type(4)))  unsigned int v4u;
typedef unsigned long long u64;

typedef __attribute__((address_space(1))) const v4u            ga_v4u;
typedef __attribute__((address_space(1))) const unsigned short ga_u16c;
typedef __attribute__((address_space(1))) unsigned short       ga_u16;
typedef __attribute__((address_space(1))) u64                  ga_u64;

__device__ __forceinline__ v4u gload4(const void* p) {
    return *(ga_v4u*)(u64)p;
}
__device__ __forceinline__ unsigned short gload_u16(const void* p) {
    return *(ga_u16c*)(u64)p;
}
__device__ __forceinline__ void gstore_u16(void* p, unsigned short v) {
    *(ga_u16*)(u64)p = v;
}
__device__ __forceinline__ void gstore_u64(void* p, u64 v) {
    *(ga_u64*)(u64)p = v;
}

__device__ __forceinline__ float bf16_bits_to_f32(unsigned int h16) {
    return __uint_as_float(h16 << 16);
}

__device__ __forceinline__ unsigned short f32_to_bf16_rne(float f) {
    unsigned int u = __float_as_uint(f);
    u += 0x7FFFu + ((u >> 16) & 1u);
    return (unsigned short)(u >> 16);
}

__device__ __forceinline__ unsigned int f32_to_e4m3(float v) {
    v = fminf(fmaxf(v, -448.0f), 448.0f);
    unsigned int u = __float_as_uint(v);
    unsigned int s = (u >> 24) & 0x80u;
    unsigned int a = u & 0x7FFFFFFFu;
    if (a <= 0x3A800000u) return s;
    if (a < 0x3C800000u) {
        unsigned int m = (unsigned int)__float2int_rn(__uint_as_float(a) * 512.0f);
        return s | m;
    }
    unsigned int e    = (a >> 23) - 127u + 7u;
    unsigned int mant = a & 0x7FFFFFu;
    unsigned int r    = mant >> 20;
    unsigned int rem  = mant & 0xFFFFFu;
    r += ((rem > 0x80000u) || ((rem == 0x80000u) && (r & 1u))) ? 1u : 0u;
    unsigned int code = (e << 3) + r;
    if (code > 0x7Eu) code = 0x7Eu;
    return s | code;
}

__global__ __launch_bounds__(256)
void amax_bf16_kernel(const v4u* __restrict__ in, int n8, unsigned int* __restrict__ amax) {
    float m = 0.0f;
    int stride = gridDim.x * blockDim.x;
    for (int i = blockIdx.x * blockDim.x + threadIdx.x; i < n8; i += stride) {
        v4u v = gload4(in + i);
        unsigned int w[4] = {v.x, v.y, v.z, v.w};
#pragma unroll
        for (int j = 0; j < 4; ++j) {
            m = fmaxf(m, fabsf(bf16_bits_to_f32(w[j] & 0xFFFFu)));
            m = fmaxf(m, fabsf(bf16_bits_to_f32(w[j] >> 16)));
        }
    }
#pragma unroll
    for (int off = 16; off > 0; off >>= 1)
        m = fmaxf(m, __shfl_xor(m, off, 32));
    __shared__ float sm[8];
    int lane = threadIdx.x & 31, wv = threadIdx.x >> 5;
    if (lane == 0) sm[wv] = m;
    __syncthreads();
    if (wv == 0) {
        m = (lane < 8) ? sm[lane] : 0.0f;
#pragma unroll
        for (int off = 4; off > 0; off >>= 1)
            m = fmaxf(m, __shfl_xor(m, off, 32));
        if (lane == 0) atomicMax(amax, __float_as_uint(m));
    }
}

__global__ __launch_bounds__(256)
void quant_swizzle_kernel(const v4u* __restrict__ in, unsigned char* __restrict__ outq,
                          int rows, int K, const unsigned int* __restrict__ amax_bits,
                          int layoutB) {
    const float am    = fmaxf(__uint_as_float(*amax_bits), 1e-12f);
    const float scale = 448.0f / am;
    const int g8    = K >> 3;
    const int kt_n  = K >> 7;
    const int total = rows * g8;
    const int stride = gridDim.x * blockDim.x;
    for (int gid = blockIdx.x * blockDim.x + threadIdx.x; gid < total; gid += stride) {
        const int row = gid / g8;
        const int gk  = gid - row * g8;
        const int kt  = gk >> 4;
        const int g   = gk & 15;
        const int rt  = row >> 4;
        const int m   = row & 15;

        v4u v = gload4(in + gid);
        unsigned int w[4] = {v.x, v.y, v.z, v.w};
        u64 q = 0;
#pragma unroll
        for (int j = 0; j < 4; ++j) {
            unsigned int b0 = f32_to_e4m3(bf16_bits_to_f32(w[j] & 0xFFFFu) * scale);
            unsigned int b1 = f32_to_e4m3(bf16_bits_to_f32(w[j] >> 16)      * scale);
            q |= (u64)(b0 | (b1 << 8)) << (16 * j);
        }

        int lane, chunk, pos;
        if (layoutB) {
            lane  = m + 16 * ((g >> 1) & 1);
            chunk = g >> 2;
            pos   = (g & 1) * 8;
        } else {
            const int j8 = (g & 1) + 2 * (g >> 2);
            lane  = m + 16 * ((g >> 1) & 1);
            chunk = j8 >> 1;
            pos   = (j8 & 1) * 8;
        }
        const size_t off = ((size_t)(rt * kt_n + kt)) * 2048 + chunk * 512 + lane * 16 + pos;
        gstore_u64(outq + off, q);
    }
}

union Frag { v16i v; v4u x4[4]; };

__global__ __launch_bounds__(256)
void fp8_wmma_gemm_kernel(const unsigned char* __restrict__ Aqs,
                          const unsigned char* __restrict__ Bqs,
                          const unsigned short* __restrict__ bias,
                          unsigned short* __restrict__ out,
                          const unsigned int* __restrict__ amax_bits,
                          int M, int N, int K)
{
    const int KT   = K >> 7;
    const int lane = threadIdx.x & 31;
    const int wave = threadIdx.x >> 5;
    const int wm   = wave >> 1;
    const int wn   = wave & 1;
    const int m0   = blockIdx.y * 128 + wm * 32;
    const int n0   = blockIdx.x * 128 + wn * 64;

    const unsigned char* aT[2];
    const unsigned char* bT[4];
#pragma unroll
    for (int i = 0; i < 2; ++i)
        aT[i] = Aqs + (size_t)(((m0 >> 4) + i) * KT) * 2048 + lane * 16;
#pragma unroll
    for (int j = 0; j < 4; ++j)
        bT[j] = Bqs + (size_t)(((n0 >> 4) + j) * KT) * 2048 + lane * 16;

    v8f acc[2][4] = {};

    for (int kt = 0; kt < KT; ++kt) {
        const int ko = kt * 2048;
        Frag a[2], b[4];
#pragma unroll
        for (int i = 0; i < 2; ++i)
#pragma unroll
            for (int c = 0; c < 4; ++c)
                a[i].x4[c] = gload4(aT[i] + ko + c * 512);
#pragma unroll
        for (int j = 0; j < 4; ++j)
#pragma unroll
            for (int c = 0; c < 4; ++c)
                b[j].x4[c] = gload4(bT[j] + ko + c * 512);
#pragma unroll
        for (int i = 0; i < 2; ++i)
#pragma unroll
            for (int j = 0; j < 4; ++j)
                acc[i][j] = __builtin_amdgcn_wmma_f32_16x16x128_fp8_fp8(
                    a[i].v, b[j].v, (short)0, acc[i][j], false, false);
    }

    const float sx = fmaxf(__uint_as_float(amax_bits[0]), 1e-12f) * (1.0f / 448.0f);
    const float sw = fmaxf(__uint_as_float(amax_bits[1]), 1e-12f) * (1.0f / 448.0f);
    const float s  = sx * sw;

    const int half = lane >> 4;
    const int l15  = lane & 15;
#pragma unroll
    for (int i = 0; i < 2; ++i) {
#pragma unroll
        for (int j = 0; j < 4; ++j) {
            const int col = n0 + j * 16 + l15;
            const float bv = bf16_bits_to_f32(gload_u16(bias + col));
#pragma unroll
            for (int r = 0; r < 8; ++r) {
                const int row = m0 + i * 16 + r + 8 * half;
                gstore_u16(out + (int64_t)row * N + col,
                           f32_to_bf16_rne(acc[i][j][r] * s + bv));
            }
        }
    }
}

extern "C" void kernel_launch(void* const* d_in, const int* in_sizes, int n_in,
                              void* d_out, int out_size, void* d_ws, size_t ws_size,
                              hipStream_t stream) {
    const unsigned short* x    = (const unsigned short*)d_in[0];
    const unsigned short* wgt  = (const unsigned short*)d_in[1];
    const unsigned short* bias = (const unsigned short*)d_in[2];
    unsigned short* out = (unsigned short*)d_out;

    const int n_x = in_sizes[0];
    const int n_w = in_sizes[1];
    const int N   = in_sizes[2];
    const int K   = n_w / N;
    const int M   = n_x / K;

    unsigned char* ws   = (unsigned char*)d_ws;
    unsigned int*  amax = (unsigned int*)ws;
    unsigned char* xq   = ws + 256;
    unsigned char* wq   = xq + (size_t)M * K;

    (void)hipMemsetAsync(amax, 0, 256, stream);

    const int threads = 256;
    const int rblocks = 2048;

    amax_bf16_kernel<<<rblocks, threads, 0, stream>>>((const v4u*)x,   n_x / 8, amax + 0);
    amax_bf16_kernel<<<rblocks, threads, 0, stream>>>((const v4u*)wgt, n_w / 8, amax + 1);

    quant_swizzle_kernel<<<rblocks, threads, 0, stream>>>((const v4u*)x,   xq, M, K, amax + 0, 0);
    quant_swizzle_kernel<<<rblocks, threads, 0, stream>>>((const v4u*)wgt, wq, N, K, amax + 1, 1);

    dim3 grid(N / 128, M / 128);
    fp8_wmma_gemm_kernel<<<grid, threads, 0, stream>>>(xq, wq, bias, out, amax, M, N, K);
}
